// IAFF_66580583022747
// MI455X (gfx1250) — matrix operand order repaired and checked by reading, not compiled, not run on hardware
//
#include <hip/hip_runtime.h>
#include <hip/hip_bf16.h>
#include <math.h>

typedef __attribute__((ext_vector_type(16))) _Float16 v16h;
typedef __attribute__((ext_vector_type(8)))  _Float16 v8h;
typedef __attribute__((ext_vector_type(2)))  _Float16 v2h;
typedef __attribute__((ext_vector_type(8)))  float    v8f;

#define EPS 1e-5f

// ---------------------------------------------------------------------------
// Weight pack: W (K=16, Cin, Cout) f32  ->  fp16 fragments in wave32 B layout.
// Fragment index: ((kc*NT + ntile)*32 + lane)*16 + j
//   k = kc*32 + (lane>>4)*16 + j ; n = ntile*16 + (lane&15)
// ---------------------------------------------------------------------------
__global__ void k_pack_w(const float* __restrict__ W, _Float16* __restrict__ pw,
                         int CIN, int COUT) {
  int KD = 16 * CIN;
  int NT = COUT >> 4;
  int total = KD * COUT;
  for (int idx = blockIdx.x * blockDim.x + threadIdx.x; idx < total;
       idx += gridDim.x * blockDim.x) {
    int j    = idx & 15;
    int lane = (idx >> 4) & 31;
    int t    = idx >> 9;
    int nt   = t % NT;
    int kc   = t / NT;
    int k = kc * 32 + ((lane >> 4) << 4) + j;
    int n = nt * 16 + (lane & 15);
    pw[idx] = (_Float16)W[k * COUT + n];
  }
}

__global__ void k_zero(float* __restrict__ p, int n) {
  int i = blockIdx.x * blockDim.x + threadIdx.x;
  if (i < n) p[i] = 0.0f;
}

// h += b ; BN (inference) ; ReLU
__global__ void k_bn_relu(const float* __restrict__ h, const float* __restrict__ b,
                          const float* __restrict__ s, const float* __restrict__ o,
                          const float* __restrict__ m, const float* __restrict__ v,
                          float* __restrict__ out, int total, int Cmask) {
  int i = blockIdx.x * blockDim.x + threadIdx.x;
  if (i >= total) return;
  int c = i & Cmask;
  float t = (h[i] + b[c] - m[c]) * rsqrtf(v[c] + EPS) * s[c] + o[c];
  out[i] = fmaxf(t, 0.0f);
}

// h += b ; BN ; sigmoid gate ; AFF fusion  out = 2x*w + 2y*(1-w)   (C = 32)
__global__ void k_fuse(const float* __restrict__ h, const float* __restrict__ b,
                       const float* __restrict__ s, const float* __restrict__ o,
                       const float* __restrict__ m, const float* __restrict__ v,
                       const float* __restrict__ x, const float* __restrict__ y,
                       float* __restrict__ out, int total) {
  int i = blockIdx.x * blockDim.x + threadIdx.x;
  if (i >= total) return;
  int c = i & 31;
  float t = (h[i] + b[c] - m[c]) * rsqrtf(v[c] + EPS) * s[c] + o[c];
  float w = 1.0f / (1.0f + __expf(-t));
  out[i] = 2.0f * x[i] * w + 2.0f * y[i] * (1.0f - w);
}

// ---------------------------------------------------------------------------
// Fused continuous-conv GEMM block.
//   EPB = 16*MT edges per block; one wave per 16-wide Cout tile.
//   g rows built in LDS (fp16, padded rows), B fragments streamed once per
//   kc from packed weights (L2 resident) and reused across MT M-tiles.
//   f32 WMMA accumulation, f32 atomic scatter-add (L2 resident).
// ---------------------------------------------------------------------------
template <int CIN, int COUT, int MT, bool CONCAT>
__global__ __launch_bounds__(32 * (COUT / 16))
void k_cconv(const float* __restrict__ featA, const float* __restrict__ featB,
             const int* __restrict__ senders, const int* __restrict__ receivers,
             const float* __restrict__ rel_pos, const float* __restrict__ a,
             const float* __restrict__ P,
             const _Float16* __restrict__ pw,
             float* __restrict__ out_acc) {
  constexpr int KD  = 16 * CIN;       // GEMM K-dimension
  constexpr int AS  = KD + 8;         // padded LDS row stride (halves)
  constexpr int NT  = COUT / 16;      // Cout tiles == waves
  constexpr int EPB = 16 * MT;        // edges per block

  __shared__ _Float16 s_g[EPB * AS];
  __shared__ float    s_w[EPB * 16];
  __shared__ int      s_snd[EPB];
  __shared__ int      s_rcv[EPB];

  const int tid = threadIdx.x;
  const int e0  = blockIdx.x * EPB;

  // ---- per-edge tap softmax (* a), edge meta ------------------------------
  for (int t = tid; t < EPB; t += blockDim.x) {
    int e = e0 + t;
    s_snd[t] = senders[e];
    s_rcv[t] = receivers[e];
    float r0 = rel_pos[e * 3 + 0];
    float r1 = rel_pos[e * 3 + 1];
    float r2 = rel_pos[e * 3 + 2];
    float lg[16];
    float mx = -1e30f;
    #pragma unroll
    for (int k = 0; k < 16; ++k) {
      float l = r0 * P[k] + r1 * P[16 + k] + r2 * P[32 + k];
      lg[k] = l;
      mx = fmaxf(mx, l);
    }
    float sum = 0.0f;
    #pragma unroll
    for (int k = 0; k < 16; ++k) {
      float t2 = __expf(lg[k] - mx);
      lg[k] = t2;
      sum += t2;
    }
    float sc = a[e] / sum;
    #pragma unroll
    for (int k = 0; k < 16; ++k) s_w[t * 16 + k] = lg[k] * sc;
  }
  __syncthreads();

  // ---- build g rows in fp16 straight from registers -----------------------
  //   g[e, tap*CIN+i] = w[e,tap] * fe[e,i]; packed 2-half stores per tap.
  for (int idx = tid; idx < EPB * (CIN / 2); idx += blockDim.x) {
    int e  = idx / (CIN / 2);
    int i  = (idx - e * (CIN / 2)) * 2;
    int sn = s_snd[e];
    float2 f2;
    if (CONCAT) {
      const float* src = (i < 32) ? &featA[sn * 32 + i] : &featB[sn * 32 + (i - 32)];
      f2 = *(const float2*)src;
    } else {
      f2 = *(const float2*)&featA[sn * CIN + i];
    }
    _Float16* grow = &s_g[e * AS + i];
    #pragma unroll
    for (int tap = 0; tap < 16; ++tap) {
      float w = s_w[e * 16 + tap];
      v2h pk = { (_Float16)(w * f2.x), (_Float16)(w * f2.y) };
      *(v2h*)(grow + tap * CIN) = pk;
    }
  }
  __syncthreads();

  // ---- WMMA GEMM: MT x (16 x KD) panels vs one (KD x 16) tile per wave ----
  const int lane = tid & 31;
  const int wv   = tid >> 5;          // which 16-wide Cout tile
  const int hi   = lane >> 4;         // lane half
  const int l15  = lane & 15;

  v8f acc[MT];
  #pragma unroll
  for (int mt = 0; mt < MT; ++mt) acc[mt] = {};

  const _Float16* pwl = pw + ((size_t)wv * 32 + lane) * 16;
  #pragma unroll 2
  for (int kc = 0; kc < KD / 32; ++kc) {
    v16h bf = *(const v16h*)(pwl + (size_t)kc * (NT * 512));
    #pragma unroll
    for (int mt = 0; mt < MT; ++mt) {
      const _Float16* ap = &s_g[(mt * 16 + l15) * AS + kc * 32 + hi * 16];
      union { v16h v; v8h h[2]; } af;
      af.h[0] = *(const v8h*)(ap);       // K = kc*32 + hi*16 .. +7
      af.h[1] = *(const v8h*)(ap + 8);   // K = kc*32 + hi*16 + 8 .. +15
      acc[mt] = __builtin_amdgcn_wmma_f32_16x16x32_f16(
          false, af.v, false, bf, (short)0, acc[mt], false, false);
    }
  }

  // ---- scatter-add to receivers (f32 atomics, L2 resident) ----------------
  #pragma unroll
  for (int mt = 0; mt < MT; ++mt) {
    #pragma unroll
    for (int r = 0; r < 8; ++r) {
      int m    = mt * 16 + r + hi * 8;  // edge row within block
      int node = s_rcv[m];
      int oc   = wv * 16 + l15;
      atomicAdd(&out_acc[node * COUT + oc], acc[mt][r]);
    }
  }
}

// ---------------------------------------------------------------------------
extern "C" void kernel_launch(void* const* d_in, const int* in_sizes, int n_in,
                              void* d_out, int out_size, void* d_ws, size_t ws_size,
                              hipStream_t stream) {
  const int C  = 32;
  const int N  = in_sizes[0] / C;     // 16384
  const int E  = in_sizes[2];         // 65536
  const int MT = 4;                   // M-tiles (edges/16) per cconv block

  const float* x        = (const float*)d_in[0];
  const float* y        = (const float*)d_in[1];
  const int*   senders  = (const int*)d_in[2];
  const int*   receivers= (const int*)d_in[3];
  const float* rel_pos  = (const float*)d_in[4];
  const float* a        = (const float*)d_in[5];
  // per layer i: W, P, b, s, o, m, v at 6 + 7*(i-1) ...
  const float* W1 = (const float*)d_in[6];  const float* P1 = (const float*)d_in[7];
  const float* b1 = (const float*)d_in[8];  const float* s1 = (const float*)d_in[9];
  const float* o1 = (const float*)d_in[10]; const float* m1 = (const float*)d_in[11];
  const float* v1 = (const float*)d_in[12];
  const float* W2 = (const float*)d_in[13]; const float* P2 = (const float*)d_in[14];
  const float* b2 = (const float*)d_in[15]; const float* s2 = (const float*)d_in[16];
  const float* o2 = (const float*)d_in[17]; const float* m2 = (const float*)d_in[18];
  const float* v2 = (const float*)d_in[19];
  const float* W3 = (const float*)d_in[20]; const float* P3 = (const float*)d_in[21];
  const float* b3 = (const float*)d_in[22]; const float* s3 = (const float*)d_in[23];
  const float* o3 = (const float*)d_in[24]; const float* m3 = (const float*)d_in[25];
  const float* v3 = (const float*)d_in[26];
  const float* W4 = (const float*)d_in[27]; const float* P4 = (const float*)d_in[28];
  const float* b4 = (const float*)d_in[29]; const float* s4 = (const float*)d_in[30];
  const float* o4 = (const float*)d_in[31]; const float* m4 = (const float*)d_in[32];
  const float* v4 = (const float*)d_in[33];

  // ---- workspace carve-out (256B aligned slices) --------------------------
  char* ws = (char*)d_ws;
  size_t off = 0;
  auto alloc = [&](size_t bytes) {
    char* p = ws + off;
    off += (bytes + 255) & ~size_t(255);
    return p;
  };
  float*    h1  = (float*)alloc((size_t)N * 64 * 4);
  float*    f1  = (float*)alloc((size_t)N * 64 * 4);
  float*    h2  = (float*)alloc((size_t)N * 32 * 4);
  float*    xo  = (float*)alloc((size_t)N * 32 * 4);
  float*    h3  = (float*)alloc((size_t)N * 64 * 4);
  float*    f3  = (float*)alloc((size_t)N * 64 * 4);
  float*    h4  = (float*)alloc((size_t)N * 32 * 4);
  _Float16* pw1 = (_Float16*)alloc((size_t)16 * 64 * 64 * 2);
  _Float16* pw2 = (_Float16*)alloc((size_t)16 * 64 * 32 * 2);
  _Float16* pw3 = (_Float16*)alloc((size_t)16 * 32 * 64 * 2);
  _Float16* pw4 = (_Float16*)alloc((size_t)16 * 64 * 32 * 2);

  const int TB = 256;
  auto blk = [&](int n) { return (n + TB - 1) / TB; };
  const int EB = E / (16 * MT);       // cconv grid

  // ---- pack weights into fp16 WMMA fragments ------------------------------
  k_pack_w<<<blk(16 * 64 * 64), TB, 0, stream>>>(W1, pw1, 64, 64);
  k_pack_w<<<blk(16 * 64 * 32), TB, 0, stream>>>(W2, pw2, 64, 32);
  k_pack_w<<<blk(16 * 32 * 64), TB, 0, stream>>>(W3, pw3, 32, 64);
  k_pack_w<<<blk(16 * 64 * 32), TB, 0, stream>>>(W4, pw4, 64, 32);

  // ---- layer 1: cconv(concat(x,y) fused) -> bn -> relu --------------------
  k_zero<<<blk(N * 64), TB, 0, stream>>>(h1, N * 64);
  k_cconv<64, 64, 4, true><<<EB, 128, 0, stream>>>(
      x, y, senders, receivers, rel_pos, a, P1, pw1, h1);
  k_bn_relu<<<blk(N * 64), TB, 0, stream>>>(h1, b1, s1, o1, m1, v1, f1, N * 64, 63);

  // ---- layer 2: cconv -> bn -> sigmoid -> AFF fuse ------------------------
  k_zero<<<blk(N * 32), TB, 0, stream>>>(h2, N * 32);
  k_cconv<64, 32, 4, false><<<EB, 64, 0, stream>>>(
      f1, nullptr, senders, receivers, rel_pos, a, P2, pw2, h2);
  k_fuse<<<blk(N * 32), TB, 0, stream>>>(h2, b2, s2, o2, m2, v2, x, y, xo, N * 32);

  // ---- layer 3: cconv(xo) -> bn -> relu -----------------------------------
  k_zero<<<blk(N * 64), TB, 0, stream>>>(h3, N * 64);
  k_cconv<32, 64, 4, false><<<EB, 128, 0, stream>>>(
      xo, nullptr, senders, receivers, rel_pos, a, P3, pw3, h3);
  k_bn_relu<<<blk(N * 64), TB, 0, stream>>>(h3, b3, s3, o3, m3, v3, f3, N * 64, 63);

  // ---- layer 4: cconv -> bn -> sigmoid -> AFF fuse -> d_out ---------------
  k_zero<<<blk(N * 32), TB, 0, stream>>>(h4, N * 32);
  k_cconv<64, 32, 4, false><<<EB, 64, 0, stream>>>(
      f3, nullptr, senders, receivers, rel_pos, a, P4, pw4, h4);
  k_fuse<<<blk(N * 32), TB, 0, stream>>>(h4, b4, s4, o4, m4, v4, x, y,
                                         (float*)d_out, N * 32);
}
